// T_GCN_3375844295142
// MI455X (gfx1250) — compile-verified, matrix operand order checked by reading, not run on hardware
//
#include <hip/hip_runtime.h>
#include <hip/hip_bf16.h>
#include <math.h>

typedef __attribute__((ext_vector_type(16))) _Float16 v16h;
typedef __attribute__((ext_vector_type(8)))  float    v8f;

#define F 32   // F_IN == F_OUT == 32

// ---------------- Graph normalization ----------------

// deg starts at 1.0 (the self-loop weight added by gcn_norm)
__global__ void init_deg(float* __restrict__ deg, int n) {
    int i = blockIdx.x * blockDim.x + threadIdx.x;
    if (i < n) deg[i] = 1.0f;
}

__global__ void deg_accum(const int* __restrict__ col, const float* __restrict__ ew,
                          float* __restrict__ deg, int e) {
    int i = blockIdx.x * blockDim.x + threadIdx.x;
    if (i < e) atomicAdd(&deg[col[i]], ew[i]);
}

__global__ void deg_to_dinv(float* __restrict__ deg, int n) {
    int i = blockIdx.x * blockDim.x + threadIdx.x;
    if (i < n) deg[i] = rsqrtf(deg[i]);   // deg >= 1 always (self-loops)
}

// ---------------- Single aggregation pass: agg = Â x ----------------

// Self-loop term doubles as the initializer of agg (no memset needed).
__global__ void agg_self(const float* __restrict__ x, const float* __restrict__ dinv,
                         float* __restrict__ agg, int n) {
    int t = blockIdx.x * blockDim.x + threadIdx.x;
    if (t < n * F) {
        float d = dinv[t >> 5];
        agg[t] = x[t] * d * d;
    }
}

// One thread per (edge, feature): coalesced 128B gather of x[row], atomic scatter to agg[col].
__global__ void agg_edges(const int* __restrict__ row, const int* __restrict__ col,
                          const float* __restrict__ ew, const float* __restrict__ dinv,
                          const float* __restrict__ x, float* __restrict__ agg, int e) {
    int t = blockIdx.x * blockDim.x + threadIdx.x;
    if (t >= e * F) return;
    int eid = t >> 5;
    int f   = t & 31;
    int r = row[eid], c = col[eid];
    float norm = dinv[r] * ew[eid] * dinv[c];
    atomicAdd(&agg[c * F + f], x[r * F + f] * norm);
}

// ---------------- Fold dense weights: Mz = Wz*Lz1, Mh = Wh*Lh1 (pre-swizzled f16 B tiles) ----
// B-operand lane layout assumed symmetric to the documented A layout:
// lane = 16*h + n (column), element e<8 -> K = 8h+e, e>=8 -> K = 16+8h+(e-8).
// 4 tiles: [Mz cols 0-15][Mz cols 16-31][Mh cols 0-15][Mh cols 16-31]
__global__ void precompute(const float* __restrict__ Wz, const float* __restrict__ bz,
                           const float* __restrict__ Wh, const float* __restrict__ bh,
                           const float* __restrict__ LzW, const float* __restrict__ Lzb,
                           const float* __restrict__ LhW, const float* __restrict__ Lhb,
                           _Float16* __restrict__ Bsw, float* __restrict__ cz,
                           float* __restrict__ ch) {
    int t = threadIdx.x;   // 192 threads, 1 block
    if (t < 128) {
        int tile = t >> 5, lane = t & 31;
        int h = lane >> 4, n = lane & 15;
        const float* W = (tile < 2) ? Wz : Wh;
        const float* L = (tile < 2) ? LzW : LhW;   // use rows 0..31 only (H==0 kills rows 32..63)
        int colj = n + ((tile & 1) << 4);
        #pragma unroll
        for (int e = 0; e < 16; ++e) {
            int k = (e < 8) ? (8 * h + e) : (16 + 8 * h + (e - 8));
            float s = 0.0f;
            #pragma unroll
            for (int i = 0; i < 32; ++i) s += W[k * 32 + i] * L[i * 32 + colj];
            Bsw[(tile * 32 + lane) * 16 + e] = (_Float16)s;
        }
    } else if (t < 160) {
        int j = t - 128; float s = Lzb[j];
        for (int i = 0; i < 32; ++i) s += bz[i] * LzW[i * 32 + j];
        cz[j] = s;
    } else if (t < 192) {
        int j = t - 160; float s = Lhb[j];
        for (int i = 0; i < 32; ++i) s += bh[i] * LhW[i * 32 + j];
        ch[j] = s;
    }
}

// ---------------- Fused GRU + readout: one wave per 16 nodes, 4 WMMAs ----------------
__global__ void gru_out(const float* __restrict__ agg, const _Float16* __restrict__ Bsw,
                        const float* __restrict__ cz, const float* __restrict__ ch,
                        const float* __restrict__ linW, const float* __restrict__ linb,
                        float* __restrict__ out, int nTiles, int n) {
    int wave = (blockIdx.x * blockDim.x + threadIdx.x) >> 5;   // uniform per wave
    int lane = threadIdx.x & 31;
    if (wave >= nTiles) return;                                // whole-wave exit; EXEC stays full

    int h = lane >> 4, m = lane & 15;
    int rowIdx = wave * 16 + m;
    if (rowIdx >= n) rowIdx = n - 1;                           // clamp (N%16==0 here anyway)

    // A tile: 16x32 f16, ISA layout for 16-bit A 16x32
    const float* arow = agg + (size_t)rowIdx * F;
    int b0 = 8 * h, b1 = 16 + 8 * h;
    v16h a;
    #pragma unroll
    for (int e = 0; e < 8; ++e) {
        a[e]     = (_Float16)arow[b0 + e];
        a[8 + e] = (_Float16)arow[b1 + e];
    }

    // Pre-swizzled B tiles: one contiguous 32B load per lane per tile
    const v16h* B = (const v16h*)Bsw;
    v16h bz0 = B[0 * 32 + lane];
    v16h bz1 = B[1 * 32 + lane];
    v16h bh0 = B[2 * 32 + lane];
    v16h bh1 = B[3 * 32 + lane];

    v8f c0 = {}, c1 = {}, c2 = {}, c3 = {};
    c0 = __builtin_amdgcn_wmma_f32_16x16x32_f16(false, a, false, bz0, (short)0, c0, false, false);
    c1 = __builtin_amdgcn_wmma_f32_16x16x32_f16(false, a, false, bz1, (short)0, c1, false, false);
    c2 = __builtin_amdgcn_wmma_f32_16x16x32_f16(false, a, false, bh0, (short)0, c2, false, false);
    c3 = __builtin_amdgcn_wmma_f32_16x16x32_f16(false, a, false, bh1, (short)0, c3, false, false);

    // C/D layout: lane -> (n0 = lane&15, rows mbase..mbase+7 with mbase = 8*(lane>>4))
    int n0 = lane & 15;
    int mbase = 8 * h;
    float czv0 = cz[n0], czv1 = cz[n0 + 16];
    float chv0 = ch[n0], chv1 = ch[n0 + 16];
    float lw0 = linW[n0], lw1 = linW[n0 + 16];
    float lb  = linb[0];

    #pragma unroll
    for (int j = 0; j < 8; ++j) {
        float z0 = 1.0f / (1.0f + expf(-(c0[j] + czv0)));
        float z1 = 1.0f / (1.0f + expf(-(c1[j] + czv1)));
        float t0 = tanhf(c2[j] + chv0);
        float t1 = tanhf(c3[j] + chv1);
        float H0 = (1.0f - z0) * t0;      // H = Z*0 + (1-Z)*H~
        float H1 = (1.0f - z1) * t1;
        float p = fmaxf(H0, 0.0f) * lw0 + fmaxf(H1, 0.0f) * lw1;
        // reduce over the 16 lanes of this half-wave (features 0..31 of row mbase+j)
        p += __shfl_xor(p, 1, 32);
        p += __shfl_xor(p, 2, 32);
        p += __shfl_xor(p, 4, 32);
        p += __shfl_xor(p, 8, 32);
        int node = wave * 16 + mbase + j;
        if (n0 == 0 && node < n) out[node] = p + lb;
    }
}

// ---------------- Launch ----------------
extern "C" void kernel_launch(void* const* d_in, const int* in_sizes, int n_in,
                              void* d_out, int out_size, void* d_ws, size_t ws_size,
                              hipStream_t stream) {
    const float* x    = (const float*)d_in[0];
    const int*   ei   = (const int*)  d_in[1];   // [2, E]
    const float* ew   = (const float*)d_in[2];
    const float* Wz   = (const float*)d_in[3];
    const float* bz   = (const float*)d_in[4];
    // d_in[5], d_in[6] (Wr, br) are mathematically dead: H==0 => H*R == 0
    const float* Wh   = (const float*)d_in[7];
    const float* bh   = (const float*)d_in[8];
    const float* LzW  = (const float*)d_in[9];
    const float* Lzb  = (const float*)d_in[10];
    // d_in[11], d_in[12] (Lr) dead as well
    const float* LhW  = (const float*)d_in[13];
    const float* Lhb  = (const float*)d_in[14];
    const float* linW = (const float*)d_in[15];
    const float* linb = (const float*)d_in[16];

    const int n = in_sizes[0] / F;     // 100000
    const int e = in_sizes[2];         // 1600000
    const int* row = ei;
    const int* col = ei + e;

    // workspace carve-up (every word read is written by a kernel first)
    char* ws = (char*)d_ws;
    float* dinv = (float*)ws;                                   // [n]  (deg, then rsqrt in place)
    float* agg  = dinv + n;                                     // [n*F]
    size_t off  = ((size_t)n * 4 + (size_t)n * F * 4 + 63) & ~(size_t)63;
    _Float16* Bsw = (_Float16*)(ws + off);                      // 4*32*16 halves = 4KB
    float* czp = (float*)(ws + off + 4 * 32 * 16 * sizeof(_Float16));
    float* chp = czp + 32;

    const int T = 256;
    init_deg   <<<(n + T - 1) / T, T, 0, stream>>>(dinv, n);
    deg_accum  <<<(e + T - 1) / T, T, 0, stream>>>(col, ew, dinv, e);
    deg_to_dinv<<<(n + T - 1) / T, T, 0, stream>>>(dinv, n);
    agg_self   <<<((n * F) + T - 1) / T, T, 0, stream>>>(x, dinv, agg, n);
    agg_edges  <<<((e * F) + T - 1) / T, T, 0, stream>>>(row, col, ew, dinv, x, agg, e);
    precompute <<<1, 192, 0, stream>>>(Wz, bz, Wh, bh, LzW, Lzb, LhW, Lhb, Bsw, czp, chp);

    int nTiles = (n + 15) / 16;        // 6250 waves, 8 waves (256 threads) per block
    int blocks = (nTiles * 32 + T - 1) / T;
    gru_out    <<<blocks, T, 0, stream>>>(agg, Bsw, czp, chp, linW, linb, (float*)d_out, nTiles, n);
}
